// LeNet5_KAN_84069689852403
// MI455X (gfx1250) — matrix operand order repaired and checked by reading, not compiled, not run on hardware
//
#include <hip/hip_runtime.h>
#include <hip/hip_bf16.h>

typedef __attribute__((ext_vector_type(16))) _Float16 v16h;
typedef __attribute__((ext_vector_type(8)))  _Float16 v8h;
typedef __attribute__((ext_vector_type(8)))  float    v8f;

#define KAN_GRID_LO (-1.0f)
#define KAN_GRID_H  (0.4f)

// Cox-de Boor cubic B-spline bases on the uniform extended knot vector
// t_i = -1 + 0.4*(i-3), i = 0..11 (matches reference _make_grid / _b_splines).
__device__ __forceinline__ void bspline8(float x, float bs[8]) {
    float b[11];
#pragma unroll
    for (int i = 0; i < 11; ++i) {
        float t0 = KAN_GRID_LO + KAN_GRID_H * (float)(i - 3);
        b[i] = (x >= t0 && x < t0 + KAN_GRID_H) ? 1.0f : 0.0f;
    }
#pragma unroll
    for (int k = 1; k <= 3; ++k) {
        float inv = 1.0f / (KAN_GRID_H * (float)k);
#pragma unroll
        for (int i = 0; i < 11 - k; ++i) {
            float ti    = KAN_GRID_LO + KAN_GRID_H * (float)(i - 3);
            float left  = (x - ti) * inv;                               // (x-t_i)/(t_{i+k}-t_i)
            float right = (ti + KAN_GRID_H * (float)(k + 1) - x) * inv; // (t_{i+k+1}-x)/(t_{i+k+1}-t_{i+1})
            b[i] = left * b[i] + right * b[i + 1];
        }
    }
#pragma unroll
    for (int j = 0; j < 8; ++j) bs[j] = b[j];
}

// ---------------------------------------------------------------------------
// Weight packing into the WMMA B-operand per-lane layout.
// B chunk (32K x 16N f16): lane L holds column N = L&15; lanes 0-15 hold
// K = chunk*32 + {0..7,16..23}, lanes 16-31 hold K = chunk*32 + {8..15,24..31}
// (matches the A layout).  Storage: bpack[(ch*32+lane)*16 + t].
// ---------------------------------------------------------------------------
__global__ void pack_kan_weights(const float* __restrict__ base_w,
                                 const float* __restrict__ spline_w,
                                 const float* __restrict__ scaler,
                                 _Float16* __restrict__ bpack,
                                 int Cout, int In, int nChunks) {
    int idx = blockIdx.x * blockDim.x + threadIdx.x;
    int total = nChunks * 512;
    if (idx >= total) return;
    int ch = idx >> 9;
    int r  = idx & 511;
    int lane = r >> 4;
    int t    = r & 15;
    int N = lane & 15;
    int K = ch * 32 + ((lane >> 4) << 3) + (t & 7) + ((t >> 3) << 4);
    int Ktot = 9 * In;
    float v = 0.0f;
    if (N < Cout && K < Ktot) {
        if (K < In) {
            v = base_w[N * In + K];                           // relu(x) path
        } else {
            int s = K - In;
            int i = s >> 3, j = s & 7;                        // spline path
            v = spline_w[(N * In + i) * 8 + j] * scaler[N * In + i];
        }
    }
    bpack[idx] = (_Float16)v;
}

__global__ void pack_fc_weights(const float* __restrict__ w,
                                _Float16* __restrict__ bpack,
                                int Nout, int Kin, int nChunks, int nColTiles) {
    int idx = blockIdx.x * blockDim.x + threadIdx.x;
    int total = nColTiles * nChunks * 512;
    if (idx >= total) return;
    int ct  = idx / (nChunks * 512);
    int rem = idx % (nChunks * 512);
    int ch = rem >> 9;
    int r  = rem & 511;
    int lane = r >> 4;
    int t    = r & 15;
    int N = ct * 16 + (lane & 15);
    int K = ch * 32 + ((lane >> 4) << 3) + (t & 7) + ((t >> 3) << 4);
    float v = (N < Nout && K < Kin) ? w[N * Kin + K] : 0.0f;
    bpack[idx] = (_Float16)v;
}

// ---------------------------------------------------------------------------
// Fused KAN convolution: per wave, build a 16-row x Kpad f16 A-tile of
// combined features (relu | spline bases) in LDS, then sweep K with
// v_wmma_f32_16x16x32_f16 against the pre-packed weight panel.
// ---------------------------------------------------------------------------
__global__ void kan_conv_wmma(const float* __restrict__ in,
                              const _Float16* __restrict__ bpack,
                              float* __restrict__ out,
                              int Bsz, int Cin, int Hin, int Win,
                              int Cout, int Ho, int Wo, int nChunks) {
    extern __shared__ _Float16 smem[];
    const int lane   = threadIdx.x & 31;
    const int wave   = threadIdx.x >> 5;
    const int nWaves = blockDim.x >> 5;
    const int Kpad = nChunks * 32;
    const int F    = Cin * 25;       // patch features per row
    const int Ktot = 9 * F;          // relu + 8 spline bases per feature
    const int HoWo = Ho * Wo;
    const int totalRows = Bsz * HoWo;
    _Float16* atile = smem + (size_t)wave * 16 * Kpad;
    const int rowBase = (blockIdx.x * nWaves + wave) * 16;

    // Pull the (tiny, L2-resident) packed weight panel toward the WGP:
    // one byte per 128B cacheline -> global_prefetch_b8.
    {
        const char* bp = (const char*)bpack;
        int lines = (nChunks * 1024) >> 7;
        for (int i = (int)threadIdx.x; i < lines; i += (int)blockDim.x)
            __builtin_prefetch(bp + (i << 7), 0, 3);
    }

    // zero K-padding columns [Ktot, Kpad)
#pragma unroll 1
    for (int r = 0; r < 16; ++r) {
        _Float16* arow = atile + r * Kpad;
        for (int k = Ktot + lane; k < Kpad; k += 32) arow[k] = (_Float16)0.0f;
    }

    // feature generation: row-outer (hoists runtime-divisor decode to 16/wave),
    // feature-inner across lanes; f/25, f%25, /5, %5 are constant-divisor.
#pragma unroll 1
    for (int r = 0; r < 16; ++r) {
        int n = rowBase + r;
        bool valid = (n < totalRows);
        int b = 0, oy = 0, ox = 0;
        if (valid) {
            b = n / HoWo;
            int rem = n % HoWo;
            oy = rem / Wo;
            ox = rem % Wo;
        }
        const float* inb = in + (size_t)b * Cin * Hin * Win;
        _Float16* arow = atile + r * Kpad;
        for (int f = lane; f < F; f += 32) {
            int c = f / 25, rest = f % 25;
            int ky = rest / 5, kx = rest % 5;
            float v = valid ? inb[(c * Hin + (oy + ky)) * Win + (ox + kx)] : 0.0f;
            arow[f] = (_Float16)fmaxf(v, 0.0f);
            float bs[8];
            bspline8(v, bs);
#pragma unroll
            for (int j = 0; j < 8; ++j) arow[F + 8 * f + j] = (_Float16)bs[j];
        }
    }
    __syncthreads();

    // WMMA K-sweep. A 16-bit layout: lane M = lane&15; halves hold
    // K = kb+{0..7} and kb+{16..23}, kb = ch*32 + 8*(lane>=16).
    v8f acc = {0.f, 0.f, 0.f, 0.f, 0.f, 0.f, 0.f, 0.f};
    const int hi = lane >> 4;
    const _Float16* arow = atile + (lane & 15) * Kpad;
    for (int ch = 0; ch < nChunks; ++ch) {
        int kb = ch * 32 + hi * 8;
        v8h alo = *(const v8h*)(arow + kb);        // 16B-aligned -> ds_load_b128
        v8h ahi = *(const v8h*)(arow + kb + 16);
        v16h a = __builtin_shufflevector(alo, ahi, 0, 1, 2, 3, 4, 5, 6, 7,
                                         8, 9, 10, 11, 12, 13, 14, 15);
        v16h bm = *(const v16h*)(bpack + ((size_t)(ch * 32 + lane)) * 16);
        acc = __builtin_amdgcn_wmma_f32_16x16x32_f16(
            false, a, false, bm, (short)0, acc, false, false);
    }

    // D layout: lanes 0-15: M=r, N=lane; lanes 16-31: M=r+8, N=lane-16
    const int col = lane & 15;
#pragma unroll
    for (int r8 = 0; r8 < 8; ++r8) {
        int n = rowBase + r8 + hi * 8;
        if (col < Cout && n < totalRows) {
            int b = n / HoWo, rem = n % HoWo;
            int oy = rem / Wo, ox = rem % Wo;
            out[((b * Cout + col) * Ho + oy) * Wo + ox] = acc[r8];
        }
    }
}

__global__ void avgpool2(const float* __restrict__ in, float* __restrict__ out,
                         int BC, int H, int W) {
    int Ho = H >> 1, Wo = W >> 1;
    int total = BC * Ho * Wo;
    int idx = blockIdx.x * blockDim.x + threadIdx.x;
    if (idx >= total) return;
    int x = idx % Wo;
    int tmp = idx / Wo;
    int y  = tmp % Ho;
    int bc = tmp / Ho;
    const float* p = in + ((size_t)bc * H + 2 * y) * W + 2 * x;
    out[idx] = 0.25f * (p[0] + p[1] + p[W] + p[W + 1]);
}

// ---------------------------------------------------------------------------
// Dense FC: one 16x16 output tile per wave, f32 activations converted to f16
// on load, bias + optional ReLU epilogue.
// ---------------------------------------------------------------------------
__global__ void fc_wmma(const float* __restrict__ A,
                        const _Float16* __restrict__ bpack,
                        const float* __restrict__ bias,
                        float* __restrict__ out,
                        int M, int Kin, int Nout, int nChunks, int nColTiles,
                        int relu) {
    const int lane   = threadIdx.x & 31;
    const int wave   = threadIdx.x >> 5;
    const int nWaves = blockDim.x >> 5;
    const int rowTiles = (M + 15) / 16;
    const int tile = blockIdx.x * nWaves + wave;
    if (tile >= rowTiles * nColTiles) return;
    const int rt = tile / nColTiles, ct = tile % nColTiles;
    const int rowBase = rt * 16;
    const int hi = lane >> 4;

    // prefetch this tile's weight panel (global_prefetch_b8)
    {
        const char* bp = (const char*)(bpack + (size_t)ct * nChunks * 512);
        int lines = (nChunks * 1024) >> 7;
        for (int i = lane; i < lines; i += 32) __builtin_prefetch(bp + (i << 7), 0, 3);
    }

    int arowIdx = rowBase + (lane & 15);
    if (arowIdx >= M) arowIdx = M - 1;   // safe clamp; tile rows guarded on store
    const float* arow = A + (size_t)arowIdx * Kin;

    v8f acc = {0.f, 0.f, 0.f, 0.f, 0.f, 0.f, 0.f, 0.f};
    for (int ch = 0; ch < nChunks; ++ch) {
        int kb = ch * 32 + hi * 8;
        v16h a;
#pragma unroll
        for (int t = 0; t < 8; ++t) {
            int k0 = kb + t, k1 = kb + 16 + t;
            a[t]     = (_Float16)(k0 < Kin ? arow[k0] : 0.0f);
            a[8 + t] = (_Float16)(k1 < Kin ? arow[k1] : 0.0f);
        }
        v16h bm = *(const v16h*)(bpack + ((size_t)((ct * nChunks + ch) * 32 + lane)) * 16);
        acc = __builtin_amdgcn_wmma_f32_16x16x32_f16(
            false, a, false, bm, (short)0, acc, false, false);
    }

    const int col = ct * 16 + (lane & 15);
#pragma unroll
    for (int r8 = 0; r8 < 8; ++r8) {
        int n = rowBase + r8 + hi * 8;
        if (n < M && col < Nout) {
            float v = acc[r8] + bias[col];
            if (relu) v = fmaxf(v, 0.0f);
            out[(size_t)n * Nout + col] = v;
        }
    }
}

// ---------------------------------------------------------------------------
extern "C" void kernel_launch(void* const* d_in, const int* in_sizes, int n_in,
                              void* d_out, int out_size, void* d_ws, size_t ws_size,
                              hipStream_t stream) {
    (void)in_sizes; (void)n_in; (void)out_size; (void)ws_size;
    const float* x      = (const float*)d_in[0];
    const float* c1_bw  = (const float*)d_in[1];
    const float* c1_sw  = (const float*)d_in[2];
    const float* c1_ss  = (const float*)d_in[3];
    const float* c2_bw  = (const float*)d_in[4];
    const float* c2_sw  = (const float*)d_in[5];
    const float* c2_ss  = (const float*)d_in[6];
    const float* fc1_w  = (const float*)d_in[7];
    const float* fc1_b  = (const float*)d_in[8];
    const float* fc2_w  = (const float*)d_in[9];
    const float* fc2_b  = (const float*)d_in[10];
    const float* fc3_w  = (const float*)d_in[11];
    const float* fc3_b  = (const float*)d_in[12];
    float* out = (float*)d_out;

    char* ws = (char*)d_ws;
    size_t off = 0;
    auto alloc = [&](size_t bytes) -> char* {
        off = (off + 255) & ~(size_t)255;
        char* p = ws + off;
        off += bytes;
        return p;
    };

    const int B = 1024;
    const int c1Chunks = 8;              // ceil(9*25 /32)
    const int c2Chunks = 43;             // ceil(9*150/32)
    const int f1Chunks = 8,  f1CT = 8;   // K=256, N=120
    const int f2Chunks = 4,  f2CT = 6;   // K=120, N=84
    const int f3Chunks = 3,  f3CT = 4;   // K=84,  N=62

    _Float16* bp1  = (_Float16*)alloc((size_t)c1Chunks * 512 * 2);
    _Float16* bp2  = (_Float16*)alloc((size_t)c2Chunks * 512 * 2);
    _Float16* bpf1 = (_Float16*)alloc((size_t)f1CT * f1Chunks * 512 * 2);
    _Float16* bpf2 = (_Float16*)alloc((size_t)f2CT * f2Chunks * 512 * 2);
    _Float16* bpf3 = (_Float16*)alloc((size_t)f3CT * f3Chunks * 512 * 2);
    float* conv1o = (float*)alloc((size_t)B * 6  * 24 * 24 * 4);
    float* pool1o = (float*)alloc((size_t)B * 6  * 12 * 12 * 4);
    float* conv2o = (float*)alloc((size_t)B * 16 * 8  * 8  * 4);
    float* pool2o = (float*)alloc((size_t)B * 256 * 4);
    float* fc1o   = (float*)alloc((size_t)B * 120 * 4);
    float* fc2o   = (float*)alloc((size_t)B * 84  * 4);

    // --- weight prep (tiny, L2-resident afterwards) ---
    pack_kan_weights<<<(c1Chunks * 512 + 255) / 256, 256, 0, stream>>>(
        c1_bw, c1_sw, c1_ss, bp1, 6, 25, c1Chunks);
    pack_kan_weights<<<(c2Chunks * 512 + 255) / 256, 256, 0, stream>>>(
        c2_bw, c2_sw, c2_ss, bp2, 16, 150, c2Chunks);
    pack_fc_weights<<<(f1CT * f1Chunks * 512 + 255) / 256, 256, 0, stream>>>(
        fc1_w, bpf1, 120, 256, f1Chunks, f1CT);
    pack_fc_weights<<<(f2CT * f2Chunks * 512 + 255) / 256, 256, 0, stream>>>(
        fc2_w, bpf2, 84, 120, f2Chunks, f2CT);
    pack_fc_weights<<<(f3CT * f3Chunks * 512 + 255) / 256, 256, 0, stream>>>(
        fc3_w, bpf3, 62, 84, f3Chunks, f3CT);

    // --- conv1: [1024,1,28,28] -> [1024,6,24,24]; 589824 rows, 4 waves/blk ---
    {
        int rows = B * 24 * 24;
        int blocks = (rows + 63) / 64;                       // 64 rows per block
        size_t shmem = (size_t)4 * 16 * (c1Chunks * 32) * 2; // 32 KB
        kan_conv_wmma<<<blocks, 128, shmem, stream>>>(
            x, bp1, conv1o, B, 1, 28, 28, 6, 24, 24, c1Chunks);
    }
    avgpool2<<<(B * 6 * 12 * 12 + 255) / 256, 256, 0, stream>>>(conv1o, pool1o, B * 6, 24, 24);

    // --- conv2: [1024,6,12,12] -> [1024,16,8,8]; 65536 rows, 1 wave/blk (43KB tile) ---
    {
        int rows = B * 8 * 8;
        int blocks = (rows + 15) / 16;                       // 16 rows per block
        size_t shmem = (size_t)1 * 16 * (c2Chunks * 32) * 2; // 44032 B
        kan_conv_wmma<<<blocks, 32, shmem, stream>>>(
            pool1o, bp2, conv2o, B, 6, 12, 12, 16, 8, 8, c2Chunks);
    }
    avgpool2<<<(B * 16 * 4 * 4 + 255) / 256, 256, 0, stream>>>(conv2o, pool2o, B * 16, 8, 8);

    // --- fully connected stack ---
    {
        int tiles = (B / 16) * f1CT; // 512 tiles
        fc_wmma<<<(tiles + 3) / 4, 128, 0, stream>>>(
            pool2o, bpf1, fc1_b, fc1o, B, 256, 120, f1Chunks, f1CT, 1);
    }
    {
        int tiles = (B / 16) * f2CT; // 384 tiles
        fc_wmma<<<(tiles + 3) / 4, 128, 0, stream>>>(
            fc1o, bpf2, fc2_b, fc2o, B, 120, 84, f2Chunks, f2CT, 1);
    }
    {
        int tiles = (B / 16) * f3CT; // 256 tiles
        fc_wmma<<<(tiles + 3) / 4, 128, 0, stream>>>(
            fc2o, bpf3, fc3_b, out, B, 84, 62, f3Chunks, f3CT, 0);
    }
}
